// Model_56770877718749
// MI455X (gfx1250) — matrix operand order repaired and checked by reading, not compiled, not run on hardware
//
#include <hip/hip_runtime.h>
#include <hip/hip_bf16.h>

typedef __attribute__((ext_vector_type(16))) _Float16 v16h;
typedef __attribute__((ext_vector_type(8)))  float    v8f;

#define LN_EPS 1e-5f
#define D_FF 2048

// ---------------------------------------------------------------------------
// Weight packing into wave32 WMMA B-operand layout (16-bit, 32x16 tile):
//   lane L: n = L&15 (column), g = L>>4 ; slot i (0..15): k = g*16 + i
// pW1: [128 ct][32 lane][16 slot]  B-tiles of ff1_w^T (K = d padded to 32)
// pW2: [64  cc][32 lane][16 slot]  B-tiles of ff2_w^T (K = 2048, N = d pad 16)
// ---------------------------------------------------------------------------
__global__ void pack_w1_kernel(const float* __restrict__ w1, _Float16* __restrict__ pW1, int d) {
    int idx = blockIdx.x * blockDim.x + threadIdx.x;      // 128*512
    if (idx >= 128 * 512) return;
    int i    = idx & 15;
    int lane = (idx >> 4) & 31;
    int ct   = idx >> 9;
    int n = lane & 15;
    int k = ((lane >> 4) << 4) + i;                        // 0..31
    float v = (k < d) ? w1[(ct * 16 + n) * d + k] : 0.0f;  // ff1_w [2048, d]
    pW1[idx] = (_Float16)v;
}

__global__ void pack_w2_kernel(const float* __restrict__ w2, _Float16* __restrict__ pW2, int d) {
    int idx = blockIdx.x * blockDim.x + threadIdx.x;      // 64*512
    if (idx >= 64 * 512) return;
    int i    = idx & 15;
    int lane = (idx >> 4) & 31;
    int cc   = idx >> 9;
    int n = lane & 15;
    int c = cc * 32 + ((lane >> 4) << 4) + i;              // global K (ff dim)
    float v = (n < d) ? w2[n * D_FF + c] : 0.0f;           // ff2_w [d, 2048]
    pW2[idx] = (_Float16)v;
}

// ---------------------------------------------------------------------------
// Stage 1: attention (head_dim==1) for the LAST token only + out-proj +
// residual + LN1.  One wave32 per batch element; lane owns tokens s, s+32.
// Writes zero-padded T[b][0..15] (f32).
// ---------------------------------------------------------------------------
template <int D>
__global__ void attn_ln1_kernel(const float* __restrict__ x,      // [B,64,D]
                                const float* __restrict__ in_w,   // [3D,D]
                                const float* __restrict__ in_b,   // [3D]
                                const float* __restrict__ out_w,  // [D,D]
                                const float* __restrict__ out_b,  // [D]
                                const float* __restrict__ ln1_w,
                                const float* __restrict__ ln1_b,
                                float* __restrict__ T)            // [B,16]
{
    const int b    = blockIdx.x;
    const int lane = threadIdx.x;                 // 0..31
    const float* xb = x + (size_t)b * 64 * D;

    float xr[2][D];
    #pragma unroll
    for (int t = 0; t < 2; ++t) {
        int s = lane + 32 * t;
        #pragma unroll
        for (int j = 0; j < D; ++j) xr[t][j] = xb[s * D + j];
    }
    // last-token row (uniform address -> scalar loads)
    float xl[D];
    #pragma unroll
    for (int j = 0; j < D; ++j) xl[j] = xb[63 * D + j];

    // q for last token (uniform)
    float q[D];
    #pragma unroll
    for (int h = 0; h < D; ++h) {
        float a = in_b[h];
        #pragma unroll
        for (int j = 0; j < D; ++j) a += xl[j] * in_w[h * D + j];
        q[h] = a;
    }
    // k,v for this lane's two tokens
    float kk[2][D], vv[2][D];
    #pragma unroll
    for (int t = 0; t < 2; ++t) {
        #pragma unroll
        for (int h = 0; h < D; ++h) {
            float ak = in_b[D + h], av = in_b[2 * D + h];
            #pragma unroll
            for (int j = 0; j < D; ++j) {
                float xv = xr[t][j];
                ak += xv * in_w[(D + h) * D + j];
                av += xv * in_w[(2 * D + h) * D + j];
            }
            kk[t][h] = ak; vv[t][h] = av;
        }
    }
    // per-head softmax over 64 tokens (scale = 1/sqrt(1) = 1)
    float ctx[D];
    #pragma unroll
    for (int h = 0; h < D; ++h) {
        float s0 = q[h] * kk[0][h], s1 = q[h] * kk[1][h];
        float m = fmaxf(s0, s1);
        for (int off = 16; off > 0; off >>= 1) m = fmaxf(m, __shfl_xor(m, off, 32));
        float e0 = expf(s0 - m), e1 = expf(s1 - m);
        float se = e0 + e1;
        float sv = e0 * vv[0][h] + e1 * vv[1][h];
        for (int off = 16; off > 0; off >>= 1) {
            se += __shfl_xor(se, off, 32);
            sv += __shfl_xor(sv, off, 32);
        }
        ctx[h] = sv / se;
    }
    // out-proj + residual + LN1 (redundant across lanes; lane 0 writes)
    float r[D]; float mu = 0.0f;
    #pragma unroll
    for (int j = 0; j < D; ++j) {
        float a = out_b[j];
        #pragma unroll
        for (int h = 0; h < D; ++h) a += ctx[h] * out_w[j * D + h];
        r[j] = xl[j] + a;
        mu += r[j];
    }
    mu /= (float)D;
    float var = 0.0f;
    #pragma unroll
    for (int j = 0; j < D; ++j) { float dl = r[j] - mu; var += dl * dl; }
    var /= (float)D;
    float inv = rsqrtf(var + LN_EPS);
    if (lane == 0) {
        float* to = T + (size_t)b * 16;
        #pragma unroll
        for (int j = 0; j < 16; ++j)
            to[j] = (j < D) ? ((r[j] - mu) * inv * ln1_w[j] + ln1_b[j]) : 0.0f;
    }
}

// ---------------------------------------------------------------------------
// Stage 2: FFN (d->2048 relu ->d) via v_wmma_f32_16x16x32_f16 + residual + LN2.
// 4 waves per 16-row batch tile; wave w owns K-chunks cc = w, w+4, ... (16 of
// 64), each with a private LDS staging slice, then partial 16x16 accumulators
// are reduced in LDS.  A layout (16-bit, 16x32):
//   lane L: m = L&15, g = L>>4 ; slot i: k = g*16 + i
// ---------------------------------------------------------------------------
template <int D>
__global__ void ffn_ln2_kernel(const float*    __restrict__ T,    // [B,16]
                               const _Float16* __restrict__ pW1,  // [128][32][16]
                               const float*    __restrict__ fb1,  // [2048]
                               const _Float16* __restrict__ pW2,  // [64][32][16]
                               const float*    __restrict__ fb2,  // [D]
                               const float*    __restrict__ ln2_w,
                               const float*    __restrict__ ln2_b,
                               float* __restrict__ Y)             // [B,16]
{
    __shared__ float hls[4][16][32];    // per-wave hidden tile (rows m, 32 ff cols)
    __shared__ float ypart[4][16][16];  // per-wave partial output tile
    const int bt   = blockIdx.x;
    const int tid  = threadIdx.x;       // 0..127
    const int wv   = tid >> 5;          // wave 0..3
    const int lane = tid & 31;
    const int n = lane & 15;
    const int g = lane >> 4;

    // activations tile in A layout (K = 16 real, 16..31 zero)
    v16h aT;
    {
        const float* trow = T + (size_t)(bt * 16 + n) * 16;
        #pragma unroll
        for (int i = 0; i < 16; ++i) {
            int k = 16 * g + i;
            aT[i] = (_Float16)((k < 16) ? trow[k] : 0.0f);
        }
    }

    v8f yacc = {};
    for (int it = 0; it < 16; ++it) {                     // 32 ff cols per iter
        const int cc = wv + 4 * it;
        __syncthreads();                                  // hls reuse guard
        // prefetch next iteration's weight tiles toward this WGP (L2-resident)
        __builtin_prefetch(pW1 + (size_t)(cc + 4) * 1024 + lane * 16, 0, 3);
        __builtin_prefetch(pW1 + (size_t)(cc + 4) * 1024 + 512 + lane * 16, 0, 3);
        __builtin_prefetch(pW2 + (size_t)(cc + 4) * 512 + lane * 16, 0, 3);
        #pragma unroll
        for (int sub = 0; sub < 2; ++sub) {
            int ct = cc * 2 + sub;
            v16h bw = *(const v16h*)(pW1 + (size_t)ct * 512 + lane * 16);
            v8f h = {};
            h = __builtin_amdgcn_wmma_f32_16x16x32_f16(false, aT, false, bw,
                                                       (short)0, h, false, false);
            float bias = fb1[ct * 16 + n];
            #pragma unroll
            for (int r = 0; r < 8; ++r) {
                float hv = fmaxf(h[r] + bias, 0.0f);      // bias + ReLU
                hls[wv][r + 8 * g][sub * 16 + n] = hv;
            }
        }
        __syncthreads();
        // gather hidden tile into A layout (in-wave transpose via LDS)
        v16h a2;
        #pragma unroll
        for (int i = 0; i < 16; ++i) {
            int k = 16 * g + i;
            a2[i] = (_Float16)hls[wv][n][k];
        }
        v16h b2v = *(const v16h*)(pW2 + (size_t)cc * 512 + lane * 16);
        yacc = __builtin_amdgcn_wmma_f32_16x16x32_f16(false, a2, false, b2v,
                                                      (short)0, yacc, false, false);
    }

    #pragma unroll
    for (int r = 0; r < 8; ++r) ypart[wv][r + 8 * g][n] = yacc[r];
    __syncthreads();

    if (tid < 16) {                                       // after all WMMAs
        const int m = tid;
        const float* trow = T + (size_t)(bt * 16 + m) * 16;
        float rr[D]; float mu = 0.0f;
        #pragma unroll
        for (int j = 0; j < D; ++j) {
            float s = ypart[0][m][j] + ypart[1][m][j] + ypart[2][m][j] + ypart[3][m][j];
            float v = trow[j] + s + fb2[j];               // residual + bias
            rr[j] = v; mu += v;
        }
        mu /= (float)D;
        float var = 0.0f;
        #pragma unroll
        for (int j = 0; j < D; ++j) { float dl = rr[j] - mu; var += dl * dl; }
        var /= (float)D;
        float inv = rsqrtf(var + LN_EPS);
        float* yo = Y + (size_t)(bt * 16 + m) * 16;
        #pragma unroll
        for (int j = 0; j < D; ++j) yo[j] = (rr[j] - mu) * inv * ln2_w[j] + ln2_b[j];
    }
}

// ---------------------------------------------------------------------------
// Stage 3: concat [B,7]+[B,10] -> 17 -> 32 -> 16 -> 8 -> 1 MLP, one thread/row
// ---------------------------------------------------------------------------
__global__ void mlp_kernel(const float* __restrict__ Y0, const float* __restrict__ Y1,
                           const float* __restrict__ w0, const float* __restrict__ b0,
                           const float* __restrict__ w1, const float* __restrict__ b1,
                           const float* __restrict__ w2, const float* __restrict__ b2,
                           const float* __restrict__ w3, const float* __restrict__ b3,
                           float* __restrict__ out, int B)
{
    int b = blockIdx.x * blockDim.x + threadIdx.x;
    if (b >= B) return;
    float y[17];
    #pragma unroll
    for (int j = 0; j < 7; ++j)  y[j]     = Y0[(size_t)b * 16 + j];
    #pragma unroll
    for (int j = 0; j < 10; ++j) y[7 + j] = Y1[(size_t)b * 16 + j];
    float h0[32];
    #pragma unroll
    for (int o = 0; o < 32; ++o) {
        float a = b0[o];
        #pragma unroll
        for (int i = 0; i < 17; ++i) a += y[i] * w0[o * 17 + i];
        h0[o] = fmaxf(a, 0.0f);
    }
    float h1[16];
    #pragma unroll
    for (int o = 0; o < 16; ++o) {
        float a = b1[o];
        #pragma unroll
        for (int i = 0; i < 32; ++i) a += h0[i] * w1[o * 32 + i];
        h1[o] = fmaxf(a, 0.0f);
    }
    float h2[8];
    #pragma unroll
    for (int o = 0; o < 8; ++o) {
        float a = b2[o];
        #pragma unroll
        for (int i = 0; i < 16; ++i) a += h1[i] * w2[o * 16 + i];
        h2[o] = fmaxf(a, 0.0f);
    }
    float a = b3[0];
    #pragma unroll
    for (int i = 0; i < 8; ++i) a += h2[i] * w3[i];
    out[b] = a;
}

// ---------------------------------------------------------------------------
extern "C" void kernel_launch(void* const* d_in, const int* in_sizes, int n_in,
                              void* d_out, int out_size, void* d_ws, size_t ws_size,
                              hipStream_t stream) {
    (void)n_in; (void)out_size; (void)ws_size;
    const float* x0 = (const float*)d_in[0];
    const float* x1 = (const float*)d_in[1];
    // enc params: base+0 in_w, +1 in_b, +2 out_w, +3 out_b, +4 ln1_w, +5 ln1_b,
    //             +6 ff1_w, +7 ff1_b, +8 ff2_w, +9 ff2_b, +10 ln2_w, +11 ln2_b
    const float* const* e0 = (const float* const*)&d_in[2];
    const float* const* e1 = (const float* const*)&d_in[14];
    const float* const* mp = (const float* const*)&d_in[26]; // w0,b0,w1,b1,w2,b2,w3,b3

    const int B = in_sizes[0] / (64 * 7);   // 4096

    // workspace carve (256B aligned)
    char* w = (char*)d_ws;
    auto carve = [&](size_t bytes) -> void* {
        void* p = (void*)w;
        w += (bytes + 255) & ~(size_t)255;
        return p;
    };
    float*    T0    = (float*)   carve((size_t)B * 16 * 4);
    float*    T1    = (float*)   carve((size_t)B * 16 * 4);
    float*    Y0    = (float*)   carve((size_t)B * 16 * 4);
    float*    Y1    = (float*)   carve((size_t)B * 16 * 4);
    _Float16* pW1_0 = (_Float16*)carve(128 * 512 * 2);
    _Float16* pW2_0 = (_Float16*)carve(64 * 512 * 2);
    _Float16* pW1_1 = (_Float16*)carve(128 * 512 * 2);
    _Float16* pW2_1 = (_Float16*)carve(64 * 512 * 2);

    // pack weights into WMMA B layout
    pack_w1_kernel<<<(128 * 512 + 255) / 256, 256, 0, stream>>>(e0[6], pW1_0, 7);
    pack_w2_kernel<<<(64 * 512 + 255) / 256, 256, 0, stream>>>(e0[8], pW2_0, 7);
    pack_w1_kernel<<<(128 * 512 + 255) / 256, 256, 0, stream>>>(e1[6], pW1_1, 10);
    pack_w2_kernel<<<(64 * 512 + 255) / 256, 256, 0, stream>>>(e1[8], pW2_1, 10);

    // stage 1: last-token attention + LN1
    attn_ln1_kernel<7><<<B, 32, 0, stream>>>(x0, e0[0], e0[1], e0[2], e0[3], e0[4], e0[5], T0);
    attn_ln1_kernel<10><<<B, 32, 0, stream>>>(x1, e1[0], e1[1], e1[2], e1[3], e1[4], e1[5], T1);

    // stage 2: WMMA FFN + LN2 (4 waves per 16-row tile)
    ffn_ln2_kernel<7><<<B / 16, 128, 0, stream>>>(T0, pW1_0, e0[7], pW2_0, e0[9], e0[10], e0[11], Y0);
    ffn_ln2_kernel<10><<<B / 16, 128, 0, stream>>>(T1, pW1_1, e1[7], pW2_1, e1[9], e1[10], e1[11], Y1);

    // stage 3: MLP head
    mlp_kernel<<<(B + 255) / 256, 256, 0, stream>>>(Y0, Y1,
        mp[0], mp[1], mp[2], mp[3], mp[4], mp[5], mp[6], mp[7],
        (float*)d_out, B);
}
